// MambaModel_76776835383637
// MI455X (gfx1250) — matrix operand order repaired and checked by reading, not compiled, not run on hardware
//
#include <hip/hip_runtime.h>
#include <hip/hip_bf16.h>

namespace {

constexpr int Bb = 2, Ls = 2048, Dm = 256, Di = 512, Ns = 16, Rr = 16, Kc = 4, Nl = 4, Do = 10;
constexpr int BL = Bb * Ls;
constexpr int XD = Rr + 2 * Ns;   // 48
constexpr float EPS = 1e-5f;

typedef __attribute__((ext_vector_type(16))) __bf16        v16bf;
typedef __attribute__((ext_vector_type(8)))  float         v8f;
typedef __attribute__((ext_vector_type(4)))  unsigned int  u32x4;

// ---------------- embedding gather ----------------
__global__ void embed_k(const int* __restrict__ ids, const float* __restrict__ emb,
                        float* __restrict__ hid) {
  int m = blockIdx.x, d = threadIdx.x;
  hid[(size_t)m * Dm + d] = emb[(size_t)ids[m] * Dm + d];
}

// ---------------- residual add + layernorm ----------------
__global__ void addln_k(const float* __restrict__ hid, float* __restrict__ resid,
                        const float* __restrict__ w, const float* __restrict__ b,
                        float* __restrict__ out, int first) {
  int m = blockIdx.x, d = threadIdx.x;
  size_t idx = (size_t)m * Dm + d;
  float r = hid[idx] + (first ? 0.0f : resid[idx]);
  resid[idx] = r;
  __shared__ float ssum[Dm], ssq[Dm];
  ssum[d] = r; ssq[d] = r * r;
  __syncthreads();
  for (int o = Dm / 2; o > 0; o >>= 1) {
    if (d < o) { ssum[d] += ssum[d + o]; ssq[d] += ssq[d + o]; }
    __syncthreads();
  }
  float mean = ssum[0] * (1.0f / Dm);
  float var  = ssq[0] * (1.0f / Dm) - mean * mean;
  out[idx] = (r - mean) * rsqrtf(var + EPS) * w[d] + b[d];
}

// ---------------- f32 -> bf16 convert ----------------
__global__ void f2bf_k(const float* __restrict__ x, __hip_bfloat16* __restrict__ y, int n) {
  int i = blockIdx.x * blockDim.x + threadIdx.x;
  if (i < n) y[i] = __float2bfloat16(x[i]);
}

__device__ __forceinline__ float epilog(float v, float bv, int epi) {
  v += bv;
  if (epi == 1) v = (v > 20.0f) ? v : log1pf(__expf(v));
  return v;
}

// ---------------- simple bf16 WMMA GEMM (one 16x16 tile per wave) ----------------
// out[M,No] = A[M,Kd] * W[No,Kd]^T (+bias); used only for x_proj (No=48)
__global__ void gemm_bf16_k(const __hip_bfloat16* __restrict__ A,
                            const __hip_bfloat16* __restrict__ W,
                            const float* __restrict__ bias,
                            float* __restrict__ out,
                            int M, int No, int Kd, int epi) {
  int wave = (blockIdx.x * blockDim.x + threadIdx.x) >> 5;
  int lane = threadIdx.x & 31;
  int tilesN = No >> 4, tilesM = M >> 4;
  int tm = wave / tilesN, tn = wave - tm * tilesN;
  if (tm >= tilesM) return;                 // wave-uniform exit: EXEC all-ones for WMMA
  int rc = lane & 15, hi = lane >> 4;
  const __hip_bfloat16* arow = A + (size_t)(tm * 16 + rc) * Kd;
  const __hip_bfloat16* wrow = W + (size_t)(tn * 16 + rc) * Kd;
  v8f acc = {};
  for (int k = 0; k < Kd; k += 32) {
    union { v16bf v; u32x4 u[2]; } af, bf;
    af.u[0] = *(const u32x4*)(arow + k + 8 * hi);
    af.u[1] = *(const u32x4*)(arow + k + 16 + 8 * hi);
    bf.u[0] = *(const u32x4*)(wrow + k + 8 * hi);
    bf.u[1] = *(const u32x4*)(wrow + k + 16 + 8 * hi);
    acc = __builtin_amdgcn_wmma_f32_16x16x32_bf16(false, af.v, false, bf.v,
                                                  (short)0, acc, false, false);
  }
  int n = tn * 16 + rc;
  float bv = bias ? bias[n] : 0.0f;
  int mbase = tm * 16 + 8 * hi;
  #pragma unroll
  for (int r = 0; r < 8; ++r)
    out[(size_t)(mbase + r) * No + n] = epilog(acc[r], bv, epi);
}

// ---------------- register-blocked bf16 WMMA GEMM ----------------
// Each wave computes a 64(M) x 32(N) macro-tile: 4 A-frags x 2 B-frags -> 8 WMMAs
// per K-step. Requires M % 64 == 0, No % 32 == 0, Kd % 32 == 0.
__global__ void gemm_bf16_t_k(const __hip_bfloat16* __restrict__ A,
                              const __hip_bfloat16* __restrict__ W,
                              const float* __restrict__ bias,
                              float* __restrict__ out,
                              int M, int No, int Kd, int epi) {
  int wave = (blockIdx.x * blockDim.x + threadIdx.x) >> 5;
  int lane = threadIdx.x & 31;
  int bigN = No >> 5;                       // 32-wide N groups
  int tm = wave / bigN, tn = wave - tm * bigN;
  if (tm * 64 >= M) return;                 // wave-uniform
  int rc = lane & 15, hi = lane >> 4;

  const __hip_bfloat16* arow[4];
  #pragma unroll
  for (int mi = 0; mi < 4; ++mi)
    arow[mi] = A + (size_t)(tm * 64 + mi * 16 + rc) * Kd;
  const __hip_bfloat16* wrow[2];
  #pragma unroll
  for (int ni = 0; ni < 2; ++ni)
    wrow[ni] = W + (size_t)(tn * 32 + ni * 16 + rc) * Kd;

  v8f acc[4][2];
  #pragma unroll
  for (int mi = 0; mi < 4; ++mi)
    #pragma unroll
    for (int ni = 0; ni < 2; ++ni) acc[mi][ni] = (v8f){};

  for (int k = 0; k < Kd; k += 32) {
    union { v16bf v; u32x4 u[2]; } af[4], bf[2];
    #pragma unroll
    for (int mi = 0; mi < 4; ++mi) {
      af[mi].u[0] = *(const u32x4*)(arow[mi] + k + 8 * hi);
      af[mi].u[1] = *(const u32x4*)(arow[mi] + k + 16 + 8 * hi);
    }
    #pragma unroll
    for (int ni = 0; ni < 2; ++ni) {
      bf[ni].u[0] = *(const u32x4*)(wrow[ni] + k + 8 * hi);
      bf[ni].u[1] = *(const u32x4*)(wrow[ni] + k + 16 + 8 * hi);
    }
    #pragma unroll
    for (int mi = 0; mi < 4; ++mi)
      #pragma unroll
      for (int ni = 0; ni < 2; ++ni)
        acc[mi][ni] = __builtin_amdgcn_wmma_f32_16x16x32_bf16(
            false, af[mi].v, false, bf[ni].v, (short)0, acc[mi][ni], false, false);
  }

  #pragma unroll
  for (int ni = 0; ni < 2; ++ni) {
    int n = tn * 32 + ni * 16 + rc;
    float bv = bias ? bias[n] : 0.0f;
    #pragma unroll
    for (int mi = 0; mi < 4; ++mi) {
      int mbase = tm * 64 + mi * 16 + 8 * hi;
      #pragma unroll
      for (int r = 0; r < 8; ++r)
        out[(size_t)(mbase + r) * No + n] = epilog(acc[mi][ni][r], bv, epi);
    }
  }
}

// ---------------- causal depthwise conv (K=4) + SiLU ----------------
__global__ void conv_silu_k(const float* __restrict__ xz, const float* __restrict__ cw,
                            const float* __restrict__ cb, float* __restrict__ xc) {
  int i = blockIdx.x * blockDim.x + threadIdx.x;   // BL*Di
  int d = i & (Di - 1);
  int m = i >> 9;                                  // /Di
  int l = m & (Ls - 1);
  float acc = cb[d];
  #pragma unroll
  for (int j = 0; j < Kc; ++j) {
    int dl = j - (Kc - 1);
    if (l + dl >= 0)
      acc += cw[d * Kc + j] * xz[(size_t)(m + dl) * (2 * Di) + d];  // xm = xz[:, :Di]
  }
  xc[i] = acc / (1.0f + __expf(-acc));
}

// ---------------- dt activation: xdbl[:, :16] -> bf16, K padded to 32 ----------------
__global__ void dtpad_act_k(const float* __restrict__ xdbl, __hip_bfloat16* __restrict__ dst) {
  int i = blockIdx.x * blockDim.x + threadIdx.x;   // BL*32
  int m = i >> 5, r = i & 31;
  float v = (r < Rr) ? xdbl[(size_t)m * XD + r] : 0.0f;
  dst[i] = __float2bfloat16(v);
}

// ---------------- dt weight [Di,16] -> bf16 [Di,32] zero-padded ----------------
__global__ void dtpad_w_k(const float* __restrict__ wsrc, __hip_bfloat16* __restrict__ dst) {
  int i = blockIdx.x * blockDim.x + threadIdx.x;   // Di*32
  int d = i >> 5, r = i & 31;
  float v = (r < Rr) ? wsrc[(size_t)d * Rr + r] : 0.0f;
  dst[i] = __float2bfloat16(v);
}

// ---------------- selective scan: parallel over (b,d), serial over L ----------------
__global__ void scan_k(const float* __restrict__ delta, const float* __restrict__ xc,
                       const float* __restrict__ xdbl, const float* __restrict__ A_log,
                       float* __restrict__ y) {
  int blk = blockIdx.x;                             // Bb * (Di/256)
  int b = blk / (Di / 256);
  int d = (blk % (Di / 256)) * 256 + threadIdx.x;
  float Am[Ns], h[Ns];
  #pragma unroll
  for (int n = 0; n < Ns; ++n) { Am[n] = -__expf(A_log[(size_t)d * Ns + n]); h[n] = 0.0f; }
  __shared__ float sBC[2 * Ns];
  for (int l = 0; l < Ls; ++l) {
    size_t m = (size_t)b * Ls + l;
    if (threadIdx.x < 2 * Ns) sBC[threadIdx.x] = xdbl[m * XD + Rr + threadIdx.x];
    __syncthreads();
    float dl = delta[m * Di + d];
    float du = dl * xc[m * Di + d];
    float acc = 0.0f;
    #pragma unroll
    for (int n = 0; n < Ns; ++n) {
      h[n] = __expf(dl * Am[n]) * h[n] + du * sBC[n];   // h = exp(dt*A)*h + dt*B*u
      acc += h[n] * sBC[Ns + n];                        // y = <h, C>
    }
    y[m * Di + d] = acc;
    __syncthreads();
  }
}

// ---------------- y = (y + Dp*xc) * silu(z) ----------------
__global__ void gate_k(float* __restrict__ y, const float* __restrict__ xc,
                       const float* __restrict__ xz, const float* __restrict__ Dp) {
  int i = blockIdx.x * blockDim.x + threadIdx.x;   // BL*Di
  int d = i & (Di - 1); int m = i >> 9;
  float z = xz[(size_t)m * (2 * Di) + Di + d];
  float yv = y[i] + Dp[d] * xc[i];
  y[i] = yv * (z / (1.0f + __expf(-z)));
}

// ---------------- mean over L ----------------
__global__ void pool_k(const float* __restrict__ hnorm, float* __restrict__ pooled) {
  int i = blockIdx.x * blockDim.x + threadIdx.x;   // Bb*Dm
  if (i >= Bb * Dm) return;
  int b = i / Dm, d = i - b * Dm;
  float s = 0.0f;
  for (int l = 0; l < Ls; ++l) s += hnorm[((size_t)b * Ls + l) * Dm + d];
  pooled[i] = s * (1.0f / Ls);
}

// ---------------- decode head [B,Do] ----------------
__global__ void dec_k(const float* __restrict__ pooled, const float* __restrict__ w,
                      const float* __restrict__ bias, float* __restrict__ out) {
  int i = threadIdx.x;
  if (i >= Bb * Do) return;
  int b = i / Do, o = i - b * Do;
  float s = bias[o];
  for (int d = 0; d < Dm; ++d) s += pooled[b * Dm + d] * w[(size_t)o * Dm + d];
  out[i] = s;
}

} // namespace

extern "C" void kernel_launch(void* const* d_in, const int* in_sizes, int n_in,
                              void* d_out, int out_size, void* d_ws, size_t ws_size,
                              hipStream_t stream) {
  (void)in_sizes; (void)n_in; (void)out_size; (void)ws_size;
  const int*   ids   = (const int*)  d_in[0];
  const float* emb   = (const float*)d_in[1];
  const float* normw = (const float*)d_in[2];
  const float* normb = (const float*)d_in[3];
  const float* inw   = (const float*)d_in[4];
  const float* cw    = (const float*)d_in[5];
  const float* cb    = (const float*)d_in[6];
  const float* xpw   = (const float*)d_in[7];
  const float* dtw   = (const float*)d_in[8];
  const float* dtb   = (const float*)d_in[9];
  const float* Alog  = (const float*)d_in[10];
  const float* Dp    = (const float*)d_in[11];
  const float* outw  = (const float*)d_in[12];
  const float* nfw   = (const float*)d_in[13];
  const float* nfb   = (const float*)d_in[14];
  const float* decw  = (const float*)d_in[15];
  const float* decb  = (const float*)d_in[16];
  float* outp = (float*)d_out;

  char* base = (char*)d_ws; size_t off = 0;
  auto alloc = [&](size_t bytes) -> void* {
    void* p = base + off;
    off += bytes; off = (off + 255) & ~(size_t)255;
    return p;
  };
  float* resid  = (float*)alloc((size_t)BL * Dm * 4);
  float* hid    = (float*)alloc((size_t)BL * Dm * 4);
  float* hnorm  = (float*)alloc((size_t)BL * Dm * 4);
  float* xz     = (float*)alloc((size_t)BL * 2 * Di * 4);
  float* xc     = (float*)alloc((size_t)BL * Di * 4);
  float* xdbl   = (float*)alloc((size_t)BL * XD * 4);
  float* delta  = (float*)alloc((size_t)BL * Di * 4);
  float* yv     = (float*)alloc((size_t)BL * Di * 4);
  float* pooled = (float*)alloc((size_t)Bb * Dm * 4);
  __hip_bfloat16* actbf = (__hip_bfloat16*)alloc((size_t)BL * Di * 2);
  __hip_bfloat16* wbf   = (__hip_bfloat16*)alloc((size_t)2 * Di * Dm * 2);
  __hip_bfloat16* dtabf = (__hip_bfloat16*)alloc((size_t)BL * 32 * 2);
  __hip_bfloat16* dtwbf = (__hip_bfloat16*)alloc((size_t)Di * 32 * 2);

  auto cvt = [&](const float* src, __hip_bfloat16* dst, int n) {
    f2bf_k<<<(n + 255) / 256, 256, 0, stream>>>(src, dst, n);
  };
  // register-blocked GEMM: 64x32 macro-tile per wave (M%64==0, No%32==0)
  auto gemm_t = [&](const __hip_bfloat16* Ab, const __hip_bfloat16* Wb, const float* bias,
                    float* out, int M, int No, int Kd, int epi) {
    int waves = (M / 64) * (No / 32);
    gemm_bf16_t_k<<<(waves + 3) / 4, 128, 0, stream>>>(Ab, Wb, bias, out, M, No, Kd, epi);
  };
  // fallback GEMM (one 16x16 tile per wave) for N not divisible by 32
  auto gemm_s = [&](const __hip_bfloat16* Ab, const __hip_bfloat16* Wb, const float* bias,
                    float* out, int M, int No, int Kd, int epi) {
    int waves = (M / 16) * (No / 16);
    gemm_bf16_k<<<(waves + 3) / 4, 128, 0, stream>>>(Ab, Wb, bias, out, M, No, Kd, epi);
  };

  embed_k<<<BL, Dm, 0, stream>>>(ids, emb, hid);

  for (int i = 0; i < Nl; ++i) {
    addln_k<<<BL, Dm, 0, stream>>>(hid, resid, normw + i * Dm, normb + i * Dm, hnorm, i == 0 ? 1 : 0);
    // in_proj: [BL,256] x [1024,256]^T -> xz [BL,1024]
    cvt(hnorm, actbf, BL * Dm);
    cvt(inw + (size_t)i * 2 * Di * Dm, wbf, 2 * Di * Dm);
    gemm_t(actbf, wbf, nullptr, xz, BL, 2 * Di, Dm, 0);
    // causal depthwise conv + SiLU -> xc [BL,512]
    conv_silu_k<<<(BL * Di) / 256, 256, 0, stream>>>(xz, cw + i * Di * Kc, cb + i * Di, xc);
    // x_proj: [BL,512] x [48,512]^T -> xdbl [BL,48]
    cvt(xc, actbf, BL * Di);
    cvt(xpw + (size_t)i * XD * Di, wbf, XD * Di);
    gemm_s(actbf, wbf, nullptr, xdbl, BL, XD, Di, 0);
    // dt_proj (+bias, softplus fused): [BL,32pad] x [512,32pad]^T -> delta [BL,512]
    dtpad_act_k<<<(BL * 32) / 256, 256, 0, stream>>>(xdbl, dtabf);
    dtpad_w_k<<<(Di * 32) / 256, 256, 0, stream>>>(dtw + (size_t)i * Di * Rr, dtwbf);
    gemm_t(dtabf, dtwbf, dtb + i * Di, delta, BL, Di, 32, 1);
    // selective scan -> yv [BL,512]
    scan_k<<<Bb * (Di / 256), 256, 0, stream>>>(delta, xc, xdbl, Alog + (size_t)i * Di * Ns, yv);
    // skip + gate
    gate_k<<<(BL * Di) / 256, 256, 0, stream>>>(yv, xc, xz, Dp + i * Di);
    // out_proj: [BL,512] x [256,512]^T -> hid [BL,256]
    cvt(yv, actbf, BL * Di);
    cvt(outw + (size_t)i * Dm * Di, wbf, Dm * Di);
    gemm_t(actbf, wbf, nullptr, hid, BL, Dm, Di, 0);
  }

  addln_k<<<BL, Dm, 0, stream>>>(hid, resid, nfw, nfb, hnorm, 0);
  pool_k<<<(Bb * Dm + 255) / 256, 256, 0, stream>>>(hnorm, pooled);
  dec_k<<<1, 32, 0, stream>>>(pooled, decw, decb, outp);
}
